// QuantumCircuitGNN_29274497090155
// MI455X (gfx1250) — matrix operand order repaired and checked by reading, not compiled, not run on hardware
//
#include <hip/hip_runtime.h>

// ---------------------------------------------------------------------------
// MI455X (gfx1250) GNN forward.
//   - Edge MLP (dominant, ~205 GFLOP) and node-update MLP run on
//     v_wmma_f32_16x16x32_bf16 (bf16 inputs, f32 accumulation).
//   - h / agg (25.6 MB each) are L2-resident on the 192 MB L2, so gather and
//     scatter-add traffic never hits HBM; HBM moves only edge_attr + indices.
//   - Everything numerics-sensitive (bias, relu boundary, LayerNorm, residual,
//     pooling, heads) stays f32.
// ---------------------------------------------------------------------------

#define NFEAT 16
#define EFEAT 8
#define GFEAT 32
#define HDIM  64
#define GTYPES 20
#define NLAYERS 4
#define NCLS 9

typedef __bf16 bf16;
typedef __attribute__((ext_vector_type(8)))  bf16  v8bf;
typedef __attribute__((ext_vector_type(16))) bf16  v16bf;
typedef __attribute__((ext_vector_type(8)))  float v8f;

static __device__ __forceinline__ bf16 tobf(float f) { return (bf16)f; }

static __device__ __forceinline__ v16bf cat16(v8bf lo, v8bf hi) {
  return __builtin_shufflevector(lo, hi, 0,1,2,3,4,5,6,7,8,9,10,11,12,13,14,15);
}

// A fragment (16 x 32 slice of a row-major bf16 tile, ld elems per row).
// 16-bit A layout: lane<16 -> row=lane, K = k0+[0..7] and k0+16+[0..7];
//                  lane>=16 -> row=lane-16, K = k0+8+[0..7] and k0+24+[0..7].
static __device__ __forceinline__ v16bf load_a_frag(const bf16* base, int ld, int k0, int lane) {
  const bf16* p = base + (lane & 15) * ld + k0 + ((lane >> 4) << 3);
  v8bf lo = *(const v8bf*)p;
  v8bf hi = *(const v8bf*)(p + 16);
  return cat16(lo, hi);
}

// B fragment from row-major W^T tile (row = output neuron, contiguous K).
// 16-bit B layout: lane<16 -> col N=lane, K=k0+[0..7] and k0+16+[0..7]; lane>=16 -> K=k0+8+[0..7] and k0+24+[0..7].
static __device__ __forceinline__ v16bf load_b_frag(const bf16* base, int ld, int k0, int lane) {
  const bf16* p = base + (lane & 15) * ld + k0 + ((lane >> 4) << 3);
  v8bf lo = *(const v8bf*)p;
  v8bf hi = *(const v8bf*)(p + 16);
  return cat16(lo, hi);
}

static __device__ __forceinline__ v8f wmma_bf16(v16bf a, v16bf b, v8f c) {
  return __builtin_amdgcn_wmma_f32_16x16x32_bf16(false, a, false, b, (short)0, c, false, false);
}

static __device__ __forceinline__ void atomicMaxF(float* addr, float val) {
  int* ai = (int*)addr;
  int cur = __float_as_int(*addr);
  while (__int_as_float(cur) < val) {
    int prev = atomicCAS(ai, cur, __float_as_int(val));
    if (prev == cur) break;
    cur = prev;
  }
}

// ---------------------------------------------------------------------------
// Weight packing: f32 [K][64] -> bf16 W^T [64][K_padded], row-major by output.
// ---------------------------------------------------------------------------
__global__ void pack_layer_kernel(const float* __restrict__ m1w, const float* __restrict__ m2w,
                                  const float* __restrict__ u1w, const float* __restrict__ u2w,
                                  const float* __restrict__ gemb,
                                  bf16* __restrict__ w1t, bf16* __restrict__ w2t,
                                  bf16* __restrict__ u1t, bf16* __restrict__ u2t,
                                  bf16* __restrict__ gbf) {
  int t = blockIdx.x * blockDim.x + threadIdx.x;
  int stride = gridDim.x * blockDim.x;
  for (int i = t; i < 64 * 160; i += stride) {          // msg1: 136 -> pad 160
    int n = i / 160, k = i % 160;
    w1t[i] = tobf(k < 136 ? m1w[k * 64 + n] : 0.0f);
  }
  for (int i = t; i < 64 * 64; i += stride) {           // msg2
    int n = i / 64, k = i % 64;
    w2t[i] = tobf(m2w[k * 64 + n]);
  }
  for (int i = t; i < 64 * 128; i += stride) {          // upd1
    int n = i / 128, k = i % 128;
    u1t[i] = tobf(u1w[k * 64 + n]);
  }
  for (int i = t; i < 64 * 64; i += stride) {           // upd2
    int n = i / 64, k = i % 64;
    u2t[i] = tobf(u2w[k * 64 + n]);
  }
  for (int i = t; i < GTYPES * 64; i += stride) gbf[i] = tobf(gemb[i]);
}

// ---------------------------------------------------------------------------
// Node embed: h = LN(relu(x @ W + b)), also emit bf16 copy of h.
// One node per 64-thread block.
// ---------------------------------------------------------------------------
__global__ void node_embed_kernel(const float* __restrict__ x, const float* __restrict__ w,
                                  const float* __restrict__ b, const float* __restrict__ lng,
                                  const float* __restrict__ lnb,
                                  float* __restrict__ h, bf16* __restrict__ hbf, int nNodes) {
  __shared__ float s[HDIM];
  __shared__ float sx[NFEAT];
  int node = blockIdx.x;
  if (node >= nNodes) return;
  int t = threadIdx.x;
  if (t < NFEAT) sx[t] = x[(long)node * NFEAT + t];
  __syncthreads();
  float acc = b[t];
  for (int k = 0; k < NFEAT; k++) acc += sx[k] * w[k * HDIM + t];
  acc = fmaxf(acc, 0.0f);
  s[t] = acc;
  __syncthreads();
  float m = 0.0f;
  for (int k = 0; k < HDIM; k++) m += s[k];
  m *= (1.0f / HDIM);
  float v = 0.0f;
  for (int k = 0; k < HDIM; k++) { float d = s[k] - m; v += d * d; }
  v *= (1.0f / HDIM);
  float r = rsqrtf(v + 1e-5f);
  float out = (acc - m) * r * lng[t] + lnb[t];
  h[(long)node * HDIM + t] = out;
  hbf[(long)node * HDIM + t] = tobf(out);
}

// ---------------------------------------------------------------------------
// Edge message MLP + scatter-add.  128 threads = 4 waves, 16 edges/wave.
//   m = relu([h[src] | gate_emb | edge_attr] @ W1 + b1) @ W2 + b2
//   agg[dst] += m        (f32 atomics, L2-resident)
// ---------------------------------------------------------------------------
__global__ void __launch_bounds__(128)
edge_msg_kernel(const bf16* __restrict__ hbf, const float* __restrict__ edge_attr,
                const int* __restrict__ src, const int* __restrict__ dst,
                const int* __restrict__ gate,
                const bf16* __restrict__ w1t, const float* __restrict__ b1,
                const bf16* __restrict__ w2t, const float* __restrict__ b2,
                const bf16* __restrict__ gbf, float* __restrict__ agg, int nEdges) {
  __shared__ bf16 sW1[64 * 160];      // 20480 B
  __shared__ bf16 sW2[64 * 64];       //  8192 B
  __shared__ bf16 sA[4][16 * 160];    // 20480 B (per-wave A tiles)
  __shared__ bf16 sH[4][16 * 64];     //  8192 B (per-wave hidden tiles)

  int t = threadIdx.x, lane = t & 31, wave = t >> 5;

  for (int i = t; i < (64 * 160 * 2) / 16; i += 128) ((int4*)sW1)[i] = ((const int4*)w1t)[i];
  for (int i = t; i < (64 * 64 * 2) / 16; i += 128)  ((int4*)sW2)[i] = ((const int4*)w2t)[i];

  int e0 = (blockIdx.x * 4 + wave) * 16;

  // Stage A tile: 16 edges x 160 bf16 (k 0..63 = h[src], 64..127 = gate emb,
  // 128..135 = edge_attr, 136..159 = zero pad).  16-byte chunks.
  for (int it = lane; it < 16 * 20; it += 32) {
    int el = it / 20, c = it % 20;
    int e = e0 + el;
    v8bf val;
#pragma unroll
    for (int j = 0; j < 8; ++j) val[j] = tobf(0.0f);
    if (e < nEdges) {
      if (c < 8) {
        int s = src[e];
        val = *(const v8bf*)(hbf + (long)s * HDIM + c * 8);
      } else if (c < 16) {
        int g = gate[e];
        val = *(const v8bf*)(gbf + g * HDIM + (c - 8) * 8);
      } else if (c == 16) {
        const float* ea = edge_attr + (long)e * EFEAT;
#pragma unroll
        for (int j = 0; j < 8; ++j) val[j] = tobf(ea[j]);
      }
    }
    *(v8bf*)(&sA[wave][el * 160 + c * 8]) = val;
  }
  __syncthreads();

  int n = lane & 15, hi4 = lane >> 4;

  // GEMM1: hidden = relu(A @ W1 + b1)  -> sH (bf16)
  for (int n0 = 0; n0 < 64; n0 += 16) {
    float bias = b1[n0 + n];
    v8f acc = {bias, bias, bias, bias, bias, bias, bias, bias};
#pragma unroll
    for (int k0 = 0; k0 < 160; k0 += 32) {
      v16bf a = load_a_frag(sA[wave], 160, k0, lane);
      v16bf b = load_b_frag(sW1 + n0 * 160, 160, k0, lane);
      acc = wmma_bf16(a, b, acc);
    }
#pragma unroll
    for (int i = 0; i < 8; ++i) {
      int m = i + hi4 * 8;  // C/D layout: VGPR i -> row i (+8 for upper lanes)
      sH[wave][m * 64 + n0 + n] = tobf(fmaxf(acc[i], 0.0f));
    }
  }
  __syncthreads();

  // GEMM2: msg = hidden @ W2 + b2, then scatter-add into agg[dst]
  for (int n0 = 0; n0 < 64; n0 += 16) {
    float bias = b2[n0 + n];
    v8f acc = {bias, bias, bias, bias, bias, bias, bias, bias};
#pragma unroll
    for (int k0 = 0; k0 < 64; k0 += 32) {
      v16bf a = load_a_frag(sH[wave], 64, k0, lane);
      v16bf b = load_b_frag(sW2 + n0 * 64, 64, k0, lane);
      acc = wmma_bf16(a, b, acc);
    }
#pragma unroll
    for (int i = 0; i < 8; ++i) {
      int m = i + hi4 * 8;
      int e = e0 + m;
      if (e < nEdges) {
        int d = dst[e];
        atomicAdd(agg + (long)d * HDIM + n0 + n, acc[i]);
      }
    }
  }
}

// ---------------------------------------------------------------------------
// Node update: u = LN(relu([h | agg] @ U1 + b1) @ U2 + b2);  h += u.
// 128 threads = 4 waves, 16 nodes/wave.
// ---------------------------------------------------------------------------
__global__ void __launch_bounds__(128)
node_upd_kernel(float* __restrict__ h, bf16* __restrict__ hbf, const float* __restrict__ agg,
                const bf16* __restrict__ u1t, const float* __restrict__ b1,
                const bf16* __restrict__ u2t, const float* __restrict__ b2,
                const float* __restrict__ lng, const float* __restrict__ lnb, int nNodes) {
  __shared__ bf16  sW1[64 * 128];     // 16384 B
  __shared__ bf16  sW2[64 * 64];      //  8192 B
  __shared__ bf16  sA[4][16 * 128];   // 16384 B (reused for hidden tile)
  __shared__ float sU[4][16 * 64];    // 16384 B

  int t = threadIdx.x, lane = t & 31, wave = t >> 5;

  for (int i = t; i < (64 * 128 * 2) / 16; i += 128) ((int4*)sW1)[i] = ((const int4*)u1t)[i];
  for (int i = t; i < (64 * 64 * 2) / 16; i += 128)  ((int4*)sW2)[i] = ((const int4*)u2t)[i];

  int r0 = (blockIdx.x * 4 + wave) * 16;

  // Stage A: 16 nodes x 128 (k 0..63 = h (bf16), 64..127 = agg (f32->bf16))
  for (int it = lane; it < 16 * 16; it += 32) {
    int rl = it / 16, c = it % 16;
    int node = r0 + rl;
    v8bf val;
#pragma unroll
    for (int j = 0; j < 8; ++j) val[j] = tobf(0.0f);
    if (node < nNodes) {
      if (c < 8) {
        val = *(const v8bf*)(hbf + (long)node * HDIM + c * 8);
      } else {
        const float* a = agg + (long)node * HDIM + (c - 8) * 8;
#pragma unroll
        for (int j = 0; j < 8; ++j) val[j] = tobf(a[j]);
      }
    }
    *(v8bf*)(&sA[wave][rl * 128 + c * 8]) = val;
  }
  __syncthreads();

  int n = lane & 15, hi4 = lane >> 4;

  // GEMM1: hidden = relu(A @ U1 + b1), kept in registers
  float hid[4][8];
  for (int n0i = 0; n0i < 4; ++n0i) {
    int n0 = n0i * 16;
    float bias = b1[n0 + n];
    v8f acc = {bias, bias, bias, bias, bias, bias, bias, bias};
#pragma unroll
    for (int k0 = 0; k0 < 128; k0 += 32) {
      v16bf a = load_a_frag(sA[wave], 128, k0, lane);
      v16bf b = load_b_frag(sW1 + n0 * 128, 128, k0, lane);
      acc = wmma_bf16(a, b, acc);
    }
#pragma unroll
    for (int i = 0; i < 8; ++i) hid[n0i][i] = fmaxf(acc[i], 0.0f);
  }
  __syncthreads();

  // Write hidden tile (bf16, [16][64]) into sA (safe: all reads of sA done)
  bf16* sHid = sA[wave];
  for (int n0i = 0; n0i < 4; ++n0i)
#pragma unroll
    for (int i = 0; i < 8; ++i) {
      int m = i + hi4 * 8;
      sHid[m * 64 + n0i * 16 + n] = tobf(hid[n0i][i]);
    }
  __syncthreads();

  // GEMM2: u = hidden @ U2 + b2  -> sU (f32)
  for (int n0 = 0; n0 < 64; n0 += 16) {
    float bias = b2[n0 + n];
    v8f acc = {bias, bias, bias, bias, bias, bias, bias, bias};
#pragma unroll
    for (int k0 = 0; k0 < 64; k0 += 32) {
      v16bf a = load_a_frag(sHid, 64, k0, lane);
      v16bf b = load_b_frag(sW2 + n0 * 64, 64, k0, lane);
      acc = wmma_bf16(a, b, acc);
    }
#pragma unroll
    for (int i = 0; i < 8; ++i) {
      int m = i + hi4 * 8;
      sU[wave][m * 64 + n0 + n] = acc[i];
    }
  }
  __syncthreads();

  // Per-row LayerNorm + residual (lanes 0..15 each own one row)
  if (lane < 16) {
    int node = r0 + lane;
    if (node < nNodes) {
      const float* u = &sU[wave][lane * 64];
      float m = 0.0f;
      for (int k = 0; k < HDIM; k++) m += u[k];
      m *= (1.0f / HDIM);
      float v = 0.0f;
      for (int k = 0; k < HDIM; k++) { float d = u[k] - m; v += d * d; }
      v *= (1.0f / HDIM);
      float r = rsqrtf(v + 1e-5f);
      for (int k = 0; k < HDIM; k++) {
        float val = (u[k] - m) * r * lng[k] + lnb[k];
        float hn = h[(long)node * HDIM + k] + val;
        h[(long)node * HDIM + k] = hn;
        hbf[(long)node * HDIM + k] = tobf(hn);
      }
    }
  }
}

// ---------------------------------------------------------------------------
// Pooling
// ---------------------------------------------------------------------------
__global__ void pool_init_kernel(float* hsum, float* hmax, float* cnt, int nGraphs) {
  int i = blockIdx.x * blockDim.x + threadIdx.x;
  if (i < nGraphs * HDIM) { hsum[i] = 0.0f; hmax[i] = -3.402823466e38f; }
  if (i < nGraphs) cnt[i] = 0.0f;
}

__global__ void pool_kernel(const float* __restrict__ h, const int* __restrict__ batch,
                            float* __restrict__ hsum, float* __restrict__ hmax,
                            float* __restrict__ cnt, int nNodes) {
  long i = (long)blockIdx.x * blockDim.x + threadIdx.x;
  if (i >= (long)nNodes * HDIM) return;
  int node = (int)(i / HDIM), k = (int)(i % HDIM);
  int b = batch[node];
  float v = h[i];
  atomicAdd(&hsum[b * HDIM + k], v);
  atomicMaxF(&hmax[b * HDIM + k], v);
  if (k == 0) atomicAdd(&cnt[b], 1.0f);
}

// ---------------------------------------------------------------------------
// Readout heads: tiny (B=64), plain VALU. One graph per 64-thread block.
// ---------------------------------------------------------------------------
__global__ void __launch_bounds__(64)
head_kernel(const float* __restrict__ gf, const float* __restrict__ hsum,
            const float* __restrict__ hmax, const float* __restrict__ cnt,
            const float* __restrict__ gw, const float* __restrict__ gb,
            const float* __restrict__ glng, const float* __restrict__ glnb,
            const float* __restrict__ c1w, const float* __restrict__ c1b,
            const float* __restrict__ clng, const float* __restrict__ clnb,
            const float* __restrict__ c2w, const float* __restrict__ c2b,
            const float* __restrict__ tsw, const float* __restrict__ tsb,
            const float* __restrict__ tow, const float* __restrict__ tob,
            const float* __restrict__ r1w, const float* __restrict__ r1b,
            const float* __restrict__ r2w, const float* __restrict__ r2b,
            float* __restrict__ out, int nGraphs) {
  __shared__ float sc[256];
  __shared__ float sd[128];
  __shared__ float se[64];
  __shared__ float sh[64];
  __shared__ float sr[32];
  int g = blockIdx.x, t = threadIdx.x;

  // global proj: g = LN(relu(gf @ Wg + bg))
  float acc = gb[t];
  for (int k = 0; k < GFEAT; k++) acc += gf[g * GFEAT + k] * gw[k * HDIM + t];
  acc = fmaxf(acc, 0.0f);
  se[t] = acc;
  __syncthreads();
  float m = 0.0f, v = 0.0f;
  for (int k = 0; k < HDIM; k++) m += se[k];
  m *= (1.0f / HDIM);
  for (int k = 0; k < HDIM; k++) { float d = se[k] - m; v += d * d; }
  v *= (1.0f / HDIM);
  float gproj = (acc - m) * rsqrtf(v + 1e-5f) * glng[t] + glnb[t];
  __syncthreads();

  // c = [mean | max | sum | g]
  float cn = fmaxf(cnt[g], 1.0f);
  float hs = hsum[g * HDIM + t], hm = hmax[g * HDIM + t];
  sc[t] = hs / cn; sc[64 + t] = hm; sc[128 + t] = hs; sc[192 + t] = gproj;
  __syncthreads();

  // comb1 (256->128) + relu, then LN
  float o0, o1;
  {
    int j0 = t * 2, j1 = t * 2 + 1;
    float a0 = c1b[j0], a1 = c1b[j1];
    for (int k = 0; k < 256; k++) { float c = sc[k]; a0 += c * c1w[k * 128 + j0]; a1 += c * c1w[k * 128 + j1]; }
    o0 = fmaxf(a0, 0.0f); o1 = fmaxf(a1, 0.0f);
    sd[j0] = o0; sd[j1] = o1;
  }
  __syncthreads();
  m = 0.0f;
  for (int k = 0; k < 128; k++) m += sd[k];
  m *= (1.0f / 128.0f);
  v = 0.0f;
  for (int k = 0; k < 128; k++) { float d = sd[k] - m; v += d * d; }
  v *= (1.0f / 128.0f);
  float rr = rsqrtf(v + 1e-5f);
  float dn0 = (o0 - m) * rr * clng[t * 2] + clnb[t * 2];
  float dn1 = (o1 - m) * rr * clng[t * 2 + 1] + clnb[t * 2 + 1];
  __syncthreads();
  sd[t * 2] = dn0; sd[t * 2 + 1] = dn1;
  __syncthreads();

  // comb2 (128->64) + relu
  float a = c2b[t];
  for (int k = 0; k < 128; k++) a += sd[k] * c2w[k * 64 + t];
  se[t] = fmaxf(a, 0.0f);
  __syncthreads();

  // thr hidden (64->64) + relu ; rt hidden (64->32) + relu
  a = tsb[t];
  for (int k = 0; k < 64; k++) a += se[k] * tsw[k * 64 + t];
  sh[t] = fmaxf(a, 0.0f);
  if (t < 32) {
    float b2 = r1b[t];
    for (int k = 0; k < 64; k++) b2 += se[k] * r1w[k * 32 + t];
    sr[t] = fmaxf(b2, 0.0f);
  }
  __syncthreads();

  if (t < NCLS - 1) {
    float o = tob[t];
    for (int k = 0; k < 64; k++) o += sh[k] * tow[k * (NCLS - 1) + t];
    out[g * (NCLS - 1) + t] = o;
  }
  if (t == 0) {
    float o = r2b[0];
    for (int k = 0; k < 32; k++) o += sr[k] * r2w[k];
    out[nGraphs * (NCLS - 1) + g] = o;
  }
}

// ---------------------------------------------------------------------------
// Host launcher.
// d_in layout (setup_inputs() dict order, depth-first over nested dicts):
//   0 x  1 edge_attr  2 global_features
//   3..6   node_embed: lin.w, lin.b, ln_g, ln_b
//   7+11*l mp[l]: gate_embed, msg1.w, msg1.b, msg2.w, msg2.b,
//                 upd1.w, upd1.b, upd2.w, upd2.b, ln_g, ln_b
//   51..54 global_proj: lin.w, lin.b, ln_g, ln_b
//   55,56 comb1.w/b  57,58 comb_ln_g/b  59,60 comb2.w/b
//   61,62 thr_shared  63,64 thr_out  65,66 rt1  67,68 rt2
//   69 edge_index [2,E]  70 edge_gate_type [E]  71 batch [N]
// ---------------------------------------------------------------------------
extern "C" void kernel_launch(void* const* d_in, const int* in_sizes, int n_in,
                              void* d_out, int out_size, void* d_ws, size_t ws_size,
                              hipStream_t stream) {
  const int nNodes  = in_sizes[0] / NFEAT;
  const int nGraphs = in_sizes[2] / GFEAT;
  const int nEdges  = in_sizes[70];

  const float* x   = (const float*)d_in[0];
  const float* ea  = (const float*)d_in[1];
  const float* gfv = (const float*)d_in[2];
  const int* ei    = (const int*)d_in[69];
  const int* gate  = (const int*)d_in[70];
  const int* batch = (const int*)d_in[71];
  const int* src = ei;
  const int* dst = ei + nEdges;

  // workspace carve-up (256B aligned)
  char* ws = (char*)d_ws;
  size_t off = 0;
  auto carve = [&](size_t bytes) -> void* {
    void* p = ws + off;
    off = (off + bytes + 255) & ~(size_t)255;
    return p;
  };
  float* h    = (float*)carve((size_t)nNodes * HDIM * sizeof(float));
  bf16*  hbf  = (bf16*)carve((size_t)nNodes * HDIM * sizeof(bf16));
  float* agg  = (float*)carve((size_t)nNodes * HDIM * sizeof(float));
  float* hsum = (float*)carve((size_t)nGraphs * HDIM * sizeof(float));
  float* hmax = (float*)carve((size_t)nGraphs * HDIM * sizeof(float));
  float* cnt  = (float*)carve((size_t)nGraphs * sizeof(float));
  bf16 *w1t[NLAYERS], *w2t[NLAYERS], *u1t[NLAYERS], *u2t[NLAYERS], *gbf[NLAYERS];
  for (int l = 0; l < NLAYERS; ++l) {
    w1t[l] = (bf16*)carve(64 * 160 * sizeof(bf16));
    w2t[l] = (bf16*)carve(64 * 64 * sizeof(bf16));
    u1t[l] = (bf16*)carve(64 * 128 * sizeof(bf16));
    u2t[l] = (bf16*)carve(64 * 64 * sizeof(bf16));
    gbf[l] = (bf16*)carve(GTYPES * 64 * sizeof(bf16));
  }

  // 1) pack weights to bf16 W^T tiles
  for (int l = 0; l < NLAYERS; ++l) {
    int pb = 7 + 11 * l;
    pack_layer_kernel<<<16, 256, 0, stream>>>(
        (const float*)d_in[pb + 1], (const float*)d_in[pb + 3],
        (const float*)d_in[pb + 5], (const float*)d_in[pb + 7],
        (const float*)d_in[pb + 0],
        w1t[l], w2t[l], u1t[l], u2t[l], gbf[l]);
  }

  // 2) node embedding
  node_embed_kernel<<<nNodes, 64, 0, stream>>>(
      x, (const float*)d_in[3], (const float*)d_in[4],
      (const float*)d_in[5], (const float*)d_in[6], h, hbf, nNodes);

  // 3) message passing layers
  int edgeBlocks = (nEdges + 63) / 64;
  int nodeBlocks = (nNodes + 63) / 64;
  for (int l = 0; l < NLAYERS; ++l) {
    int pb = 7 + 11 * l;
    hipMemsetAsync(agg, 0, (size_t)nNodes * HDIM * sizeof(float), stream);
    edge_msg_kernel<<<edgeBlocks, 128, 0, stream>>>(
        hbf, ea, src, dst, gate,
        w1t[l], (const float*)d_in[pb + 2],
        w2t[l], (const float*)d_in[pb + 4],
        gbf[l], agg, nEdges);
    node_upd_kernel<<<nodeBlocks, 128, 0, stream>>>(
        h, hbf, agg,
        u1t[l], (const float*)d_in[pb + 6],
        u2t[l], (const float*)d_in[pb + 8],
        (const float*)d_in[pb + 9], (const float*)d_in[pb + 10], nNodes);
  }

  // 4) pooling
  pool_init_kernel<<<(nGraphs * HDIM + 255) / 256, 256, 0, stream>>>(hsum, hmax, cnt, nGraphs);
  long poolThreads = (long)nNodes * HDIM;
  pool_kernel<<<(int)((poolThreads + 255) / 256), 256, 0, stream>>>(h, batch, hsum, hmax, cnt, nNodes);

  // 5) heads
  head_kernel<<<nGraphs, 64, 0, stream>>>(
      gfv, hsum, hmax, cnt,
      (const float*)d_in[51], (const float*)d_in[52], (const float*)d_in[53], (const float*)d_in[54],
      (const float*)d_in[55], (const float*)d_in[56], (const float*)d_in[57], (const float*)d_in[58],
      (const float*)d_in[59], (const float*)d_in[60],
      (const float*)d_in[61], (const float*)d_in[62], (const float*)d_in[63], (const float*)d_in[64],
      (const float*)d_in[65], (const float*)d_in[66], (const float*)d_in[67], (const float*)d_in[68],
      (float*)d_out, nGraphs);
}
